// ProtoModule_50474455662928
// MI455X (gfx1250) — matrix operand order repaired and checked by reading, not compiled, not run on hardware
//
#include <hip/hip_runtime.h>
#include <hip/hip_bf16.h>
#include <math.h>

typedef __attribute__((ext_vector_type(16))) _Float16 v16h;
typedef __attribute__((ext_vector_type(8)))  _Float16 v8h;
typedef __attribute__((ext_vector_type(8)))  float    v8f;

#define C_DIM   512
#define HW      4096
#define NPROTO  1024
#define TEMP    20.0f
#define THRESH  0.95f
#define EPS     1e-4f

// ---------------------------------------------------------------------------
// Kernel 1: one block per prototype cell (b, gy, gx).
// 8x8 avg-pool of sup_x, L2-normalize the row, emit f16 prototype row-major
// [p][512] (WMMA-B friendly), plus the valid flag and the proto_grid output.
// ---------------------------------------------------------------------------
__global__ __launch_bounds__(256)
void proto_kernel(const float* __restrict__ sup_x, const float* __restrict__ sup_y,
                  _Float16* __restrict__ protosB, float* __restrict__ validArr,
                  float* __restrict__ out_grid) {
    __shared__ float smean[C_DIM];
    __shared__ float red[256];
    __shared__ float sInv;

    const int p    = blockIdx.x;          // 0..1023
    const int b    = p >> 6;
    const int cell = p & 63;
    const int gy   = cell >> 3;
    const int gx   = cell & 7;
    const int t    = threadIdx.x;

    const float* xb = sup_x + (size_t)b * C_DIM * HW;
    float ss = 0.0f;
#pragma unroll
    for (int j = 0; j < 2; ++j) {
        const int c = t + j * 256;
        float s = 0.0f;
        for (int iy = 0; iy < 8; ++iy) {
            const float* row = xb + (size_t)c * HW + (gy * 8 + iy) * 64 + gx * 8;
#pragma unroll
            for (int ix = 0; ix < 8; ++ix) s += row[ix];
        }
        const float m = s * (1.0f / 64.0f);
        smean[c] = m;
        ss += m * m;
    }
    red[t] = ss;
    __syncthreads();
    for (int off = 128; off > 0; off >>= 1) {
        if (t < off) red[t] += red[t + off];
        __syncthreads();
    }
    if (t == 0) {
        sInv = 1.0f / fmaxf(sqrtf(red[0]), EPS);
        const float* yb = sup_y + (size_t)b * HW;
        float ys = 0.0f;
        for (int iy = 0; iy < 8; ++iy)
            for (int ix = 0; ix < 8; ++ix)
                ys += yb[(gy * 8 + iy) * 64 + gx * 8 + ix];
        const float g = ys * (1.0f / 64.0f);
        validArr[p] = (g > THRESH) ? 1.0f : 0.0f;
        out_grid[p] = (g < THRESH) ? 0.0f : g;
    }
    __syncthreads();
    const float inv = sInv;
#pragma unroll
    for (int j = 0; j < 2; ++j) {
        const int c = t + j * 256;
        protosB[(size_t)p * C_DIM + c] = (_Float16)(smean[c] * inv);
    }
}

// ---------------------------------------------------------------------------
// Kernel 2: per-pixel inverse channel norm of qry (coalesced over HW).
// ---------------------------------------------------------------------------
__global__ __launch_bounds__(256)
void qnorm_kernel(const float* __restrict__ qry, float* __restrict__ invn) {
    const int g  = blockIdx.x * 256 + threadIdx.x;   // 0..65535
    const int b  = g >> 12;
    const int hw = g & (HW - 1);
    const float* base = qry + (size_t)b * C_DIM * HW + hw;
    float s = 0.0f;
    for (int c = 0; c < C_DIM; ++c) {
        const float v = base[(size_t)c * HW];
        s += v * v;
    }
    invn[g] = 1.0f / fmaxf(sqrtf(s), EPS);
}

// ---------------------------------------------------------------------------
// Kernel 3: fused GEMM + online softmax + argmax.
// Block = 256 threads (8 waves), 128 pixels. A tile staged in 128KB dynamic
// LDS as f16 [pixel][K]; each wave owns 16 pixels, holds all 16 A K-fragments
// plus a DOUBLE-BUFFERED 16-fragment B tile in registers, and sweeps 64
// N-tiles of 16 prototypes with v_wmma_f32_16x16x32_f16. B loads for tile
// nt+1 are issued before computing tile nt, so the s_wait before the WMMAs
// covers loads a full tile old (latency hidden behind 16 WMMAs + epilogue).
// ---------------------------------------------------------------------------
__global__ __launch_bounds__(256)
void gemm_softmax_kernel(const float* __restrict__ qry, const float* __restrict__ invn,
                         const _Float16* __restrict__ protosB,
                         const float* __restrict__ validArr,
                         float* __restrict__ pred, float* __restrict__ dbg) {
    extern __shared__ _Float16 Alds[];      // 128 * 512 f16 = 128 KB (dynamic)

    const int tid  = threadIdx.x;
    const int base = blockIdx.x * 128;      // global pixel base (tiles never cross b)
    const int b    = base >> 12;
    const int hw0  = base & (HW - 1);

    // ---- stage A tile: normalize-on-load, transpose [C,HW] -> [pixel][K] f16
    const int px = tid & 127;
    const int cp = tid >> 7;                // 0/1 -> two channels per iteration
    const float inv = invn[base + px];
    const float* qbase = qry + (size_t)b * C_DIM * HW + hw0 + px;
    for (int c0 = 0; c0 < C_DIM; c0 += 2) {
        const int c = c0 + cp;
        Alds[px * C_DIM + c] = (_Float16)(qbase[(size_t)c * HW] * inv);
    }
    __syncthreads();

    const int wave = tid >> 5;              // 0..7
    const int lane = tid & 31;
    const int lh   = lane & 15;
    const int hi   = lane >> 4;             // 0: K-low half, 1: K-high half
    const int m0   = wave * 16;
    const int m    = m0 + lh;               // A row this lane supplies

    // ---- preload all 16 A K-fragments into registers (contiguous K order: lane half hi holds k = 16*hi + s)
    v16h afrag[16];
#pragma unroll
    for (int kc = 0; kc < 16; ++kc) {
        afrag[kc] = *(const v16h*)&Alds[m * C_DIM + kc * 32 + hi * 16];
    }

    // per-pixel online softmax state: this lane covers M = r + 8*hi, N = lh (mod 16)
    float mx[8], lsum[8], ssum[8], pbest[8];
#pragma unroll
    for (int r = 0; r < 8; ++r) { mx[r] = -1e30f; lsum[r] = 0.0f; ssum[r] = 0.0f; pbest[r] = 0.0f; }

    // B-tile loader: 16 fragments (32 x global_load_b128) into a register buffer
    auto loadB = [&](int nt, v16h* dst) {
        const _Float16* bptr = protosB + (size_t)(nt * 16 + lh) * C_DIM + hi * 16;
#pragma unroll
        for (int kc = 0; kc < 16; ++kc)
            dst[kc] = *(const v16h*)(bptr + kc * 32);
    };

    // one N-tile: 16 back-to-back WMMAs + online softmax update
    auto tile = [&](int nt, const v16h* bf) {
        v8f acc = {};
#pragma unroll
        for (int kc = 0; kc < 16; ++kc)
            acc = __builtin_amdgcn_wmma_f32_16x16x32_f16(
                false, afrag[kc], false, bf[kc], (short)0, acc, false, false);
        const int n = nt * 16 + lh;
        const float vflag = validArr[n];
        if (vflag > 0.5f) {
#pragma unroll
            for (int r = 0; r < 8; ++r) {
                const float d = acc[r] * TEMP;
                if (d > mx[r]) {
                    const float sc = __expf(mx[r] - d);
                    lsum[r] = lsum[r] * sc + 1.0f;
                    ssum[r] = ssum[r] * sc + d;
                    mx[r] = d;
                    pbest[r] = (float)n;
                } else {
                    const float e = __expf(d - mx[r]);
                    lsum[r] += e;
                    ssum[r] += e * d;
                }
            }
        }
    };

    v16h bbuf0[16], bbuf1[16];
    loadB(0, bbuf0);
    for (int nt = 0; nt < NPROTO / 16; nt += 2) {
        loadB(nt + 1, bbuf1);               // prefetch odd tile
        tile(nt, bbuf0);                    // compute even tile
        if (nt + 2 < NPROTO / 16)
            loadB(nt + 2, bbuf0);           // prefetch next even tile
        tile(nt + 1, bbuf1);                // compute odd tile
    }

    // ---- merge softmax states across the 16 lanes sharing each pixel set
#pragma unroll
    for (int off = 1; off < 16; off <<= 1) {
#pragma unroll
        for (int r = 0; r < 8; ++r) {
            const float m2 = __shfl_xor(mx[r],    off, 32);
            const float l2 = __shfl_xor(lsum[r],  off, 32);
            const float s2 = __shfl_xor(ssum[r],  off, 32);
            const float i2 = __shfl_xor(pbest[r], off, 32);
            const float M  = fmaxf(mx[r], m2);
            const float sa = __expf(mx[r] - M);
            const float sb = __expf(m2 - M);
            lsum[r] = lsum[r] * sa + l2 * sb;
            ssum[r] = ssum[r] * sa + s2 * sb;
            pbest[r] = (m2 > mx[r]) ? i2 : ((mx[r] > m2) ? pbest[r] : fminf(pbest[r], i2));
            mx[r] = M;
        }
    }

    if (lh == 0) {
#pragma unroll
        for (int r = 0; r < 8; ++r) {
            const int g = base + m0 + hi * 8 + r;
            pred[g] = (lsum[r] > 0.0f) ? (ssum[r] / lsum[r]) : 0.0f;
            dbg[g]  = pbest[r];
        }
    }
}

// ---------------------------------------------------------------------------
extern "C" void kernel_launch(void* const* d_in, const int* in_sizes, int n_in,
                              void* d_out, int out_size, void* d_ws, size_t ws_size,
                              hipStream_t stream) {
    const float* qry   = (const float*)d_in[0];   // [16,512,64,64]
    const float* sup_x = (const float*)d_in[1];   // [16,512,64,64]
    const float* sup_y = (const float*)d_in[2];   // [16,1,64,64]

    float* out  = (float*)d_out;
    float* pred = out;               // [16,1,64,64]  -> 65536
    float* dbg  = out + 65536;       // [16,64,64]    -> 65536
    float* grid = out + 131072;      // [16,1,8,8]    -> 1024

    char* ws = (char*)d_ws;
    _Float16* protosB  = (_Float16*)ws;                                // 1 MB
    float*    validArr = (float*)(ws + (size_t)NPROTO * C_DIM * 2);    // 4 KB
    float*    invn     = (float*)(ws + (size_t)NPROTO * C_DIM * 2 + NPROTO * 4); // 256 KB

    proto_kernel<<<NPROTO, 256, 0, stream>>>(sup_x, sup_y, protosB, validArr, grid);
    qnorm_kernel<<<65536 / 256, 256, 0, stream>>>(qry, invn);

    const size_t ldsBytes = (size_t)128 * C_DIM * sizeof(_Float16);    // 128 KB
    gemm_softmax_kernel<<<65536 / 128, 256, ldsBytes, stream>>>(
        qry, invn, protosB, validArr, pred, dbg);
}
